// GraphAttention_25245817766129
// MI455X (gfx1250) — matrix operand order repaired and checked by reading, not compiled, not run on hardware
//
#include <hip/hip_runtime.h>
#include <stdint.h>

// GAT layer: N=4096 nodes, F=512 in-feats, FH=64 per-head out, H=8 heads.
//   K0a/K0b: convert X -> f16, W -> f16 transposed [H][FH][F]
//   K1: feats = X @ W per head via v_wmma_f32_16x16x32_f16
//   K2: s_self/s_neigh per (h,n)
//   K3: per-row softmax stats (exact full-row softmax incl. exp(0) terms)
//   K4: out = softmax(dense) @ feats; P built on the fly; B-panel (featsT
//       tile shared by all 4 waves of the block) streamed into LDS by the
//       Tensor Data Mover (tensor_load_to_lds, 6-arg clang-23 form),
//       double-buffered with s_wait_tensorcnt; fragments read via ds_load_b128.

#define NN   4096
#define FIN  512
#define FH   64
#define HH   8
#define ALPHA 0.2f

typedef __attribute__((ext_vector_type(16))) _Float16 v16h;
typedef __attribute__((ext_vector_type(8)))  float    v8f;
typedef __attribute__((ext_vector_type(4)))  float    v4f;
typedef __attribute__((ext_vector_type(4)))  uint32_t v4u;
typedef __attribute__((ext_vector_type(8)))  int      v8i;
typedef __attribute__((ext_vector_type(4)))  int      v4i;

union FragAB {
  v4u      q[2];   // two 16B chunks
  v16h     v;
  _Float16 h[16];
};

// LDS row stride for the TDM-padded B panel: 32 f16 = 64B data + 16B pad.
#define BPANEL_STRIDE 80
#define BPANEL_BYTES  (64 * BPANEL_STRIDE)   // 5120 B per buffer

// ---- TDM: 2D tile load Global -> LDS (6-arg builtin on this toolchain) ----
// D# per CDNA5 ISA 8.3/8.4. data_size=2B, pad_enable, pad_interval=16 DW
// (pad after each 64B row), pad_amount=4 DW (16B) -> 80B LDS row stride.
__device__ __forceinline__ void tdm_load_tile_2d(uint32_t lds_addr,
                                                 const void* gptr,
                                                 uint32_t tensor_d0,
                                                 uint32_t tensor_d1,
                                                 uint32_t tile_d0,
                                                 uint32_t tile_d1,
                                                 uint64_t stride0) {
  uint64_t ga = (uint64_t)(uintptr_t)gptr;
  v4u g0;
  g0[0] = 1u;                                              // count=1 (valid D#)
  g0[1] = lds_addr;                                        // LDS byte address
  g0[2] = (uint32_t)ga;                                    // global_addr[31:0]
  g0[3] = (uint32_t)((ga >> 32) & 0x01FFFFFFu) | (2u << 30); // [120:96] + type=2
  v8i g1;
  g1[0] = (int)((1u << 16) |          // data_size = 2 bytes
                (1u << 20) |          // pad_enable
                (3u << 22) |          // pad_interval: 16 DWORDs (64B)
                (3u << 25));          // pad_amount: 4 DWORDs (16B)
  g1[1] = (int)((tensor_d0 & 0xFFFFu) << 16);              // tensor_dim0 lo16
  g1[2] = (int)((tensor_d0 >> 16) | ((tensor_d1 & 0xFFFFu) << 16));
  g1[3] = (int)((tensor_d1 >> 16) | (tile_d0 << 16));      // tile_dim0
  g1[4] = (int)tile_d1;                                    // tile_dim1, tile_dim2=0
  g1[5] = (int)(uint32_t)stride0;                          // dim0 stride lo32
  g1[6] = (int)(uint32_t)(stride0 >> 32);                  // dim0 stride hi16
  g1[7] = 0;
  v4i z4 = {0, 0, 0, 0};                                   // 2D: groups 2/3 unused
  v8i z8 = {0, 0, 0, 0, 0, 0, 0, 0};
  __builtin_amdgcn_tensor_load_to_lds(g0, g1, z4, z4, z8, 0);
}

// ---------------- conversion kernels ----------------

__global__ void k_cvt_x(const float* __restrict__ X, _Float16* __restrict__ Xh, int n) {
  int i = blockIdx.x * blockDim.x + threadIdx.x;
  if (i < n) Xh[i] = (_Float16)X[i];
}

// W in [H][F][FH] -> Wt f16 [H][FH][F]
__global__ void k_cvt_w(const float* __restrict__ W, _Float16* __restrict__ Wt) {
  int i = blockIdx.x * blockDim.x + threadIdx.x;
  if (i >= HH * FIN * FH) return;
  int h = i / (FIN * FH);
  int r = i % (FIN * FH);
  int f = r / FH, o = r % FH;
  Wt[((size_t)(h * FH + o)) * FIN + f] = (_Float16)W[i];
}

// ---------------- K1: projection GEMM (per head) ----------------
__global__ void __launch_bounds__(128) k_proj(const _Float16* __restrict__ Xh,
                                              const _Float16* __restrict__ Wt,
                                              float*    __restrict__ featsF,
                                              _Float16* __restrict__ featsT) {
  int wid  = blockIdx.x * 4 + (threadIdx.x >> 5);
  int h    = wid >> 8;
  int mt   = wid & 255;
  int lane = threadIdx.x & 31;
  int half = lane >> 4;
  int lr   = lane & 15;

  v8f acc[4];
#pragma unroll
  for (int t = 0; t < 4; ++t)
#pragma unroll
    for (int r = 0; r < 8; ++r) acc[t][r] = 0.0f;

  const _Float16* xrow = Xh + (size_t)(mt * 16 + lr) * FIN;

  for (int kb = 0; kb < FIN; kb += 32) {
    FragAB a;
    a.q[0] = *(const v4u*)(xrow + kb + half * 8);
    a.q[1] = *(const v4u*)(xrow + kb + 16 + half * 8);
#pragma unroll
    for (int t = 0; t < 4; ++t) {
      const _Float16* wp = Wt + ((size_t)(h * FH + t * 16 + lr)) * FIN + kb + half * 8;
      FragAB b;
      b.q[0] = *(const v4u*)(wp);
      b.q[1] = *(const v4u*)(wp + 16);
      acc[t] = __builtin_amdgcn_wmma_f32_16x16x32_f16(
          false, a.v, false, b.v, (short)0, acc[t], false, false);
    }
  }

#pragma unroll
  for (int t = 0; t < 4; ++t) {
    int n = t * 16 + lr;
#pragma unroll
    for (int r = 0; r < 8; ++r) {
      int M = mt * 16 + half * 8 + r;
      float v = acc[t][r];
      featsF[((size_t)h * NN + M) * FH + n] = v;
      featsT[((size_t)(h * FH + n)) * NN + M] = (_Float16)v;
    }
  }
}

// ---------------- K2: attention scores ----------------
__global__ void k_scores(const float* __restrict__ featsF,
                         const float* __restrict__ a_self,
                         const float* __restrict__ a_neigh,
                         float* __restrict__ sSelf, float* __restrict__ sNeigh) {
  int idx = blockIdx.x * blockDim.x + threadIdx.x;
  if (idx >= HH * NN) return;
  int h = idx >> 12, n = idx & (NN - 1);
  const float* fp = featsF + ((size_t)h * NN + n) * FH;
  const float* as = a_self  + h * FH;
  const float* an = a_neigh + h * FH;
  float ss = 0.f, sn = 0.f;
#pragma unroll 4
  for (int o = 0; o < FH; ++o) { float v = fp[o]; ss += v * as[o]; sn += v * an[o]; }
  sSelf[idx] = ss; sNeigh[idx] = sn;
}

// ---------------- K3: per-row softmax statistics ----------------
__global__ void __launch_bounds__(256) k_stats(const float* __restrict__ A,
                                               const float* __restrict__ sSelf,
                                               const float* __restrict__ sNeigh,
                                               float* __restrict__ rowM,
                                               float* __restrict__ rowInvZ,
                                               float* __restrict__ rowENM) {
  __shared__ float red[256];
  __shared__ float mh[HH];
  __shared__ int   degsh;
  int i   = blockIdx.x;
  int tid = threadIdx.x;
  const float* arow = A + (size_t)i * NN;

  float ss[HH];
#pragma unroll
  for (int h = 0; h < HH; ++h) ss[h] = sSelf[h * NN + i];

  float mx[HH];
#pragma unroll
  for (int h = 0; h < HH; ++h) mx[h] = 0.0f;
  int deg = 0;

  for (int j = tid; j < NN; j += 256) {
    float a = arow[j];
    if (a > 0.5f) {
      deg++;
#pragma unroll
      for (int h = 0; h < HH; ++h) {
        float l = ss[h] + sNeigh[h * NN + j];
        l = l > 0.f ? l : ALPHA * l;
        mx[h] = fmaxf(mx[h], l);
      }
    }
  }

  red[tid] = (float)deg;
  __syncthreads();
  for (int s = 128; s > 0; s >>= 1) { if (tid < s) red[tid] += red[tid + s]; __syncthreads(); }
  if (tid == 0) degsh = (int)red[0];
  __syncthreads();

  for (int h = 0; h < HH; ++h) {
    red[tid] = mx[h];
    __syncthreads();
    for (int s = 128; s > 0; s >>= 1) { if (tid < s) red[tid] = fmaxf(red[tid], red[tid + s]); __syncthreads(); }
    if (tid == 0) mh[h] = red[0];
    __syncthreads();
  }

  float mloc[HH], sm[HH];
#pragma unroll
  for (int h = 0; h < HH; ++h) { mloc[h] = mh[h]; sm[h] = 0.f; }

  for (int j = tid; j < NN; j += 256) {
    float a = arow[j];
    if (a > 0.5f) {
#pragma unroll
      for (int h = 0; h < HH; ++h) {
        float l = ss[h] + sNeigh[h * NN + j];
        l = l > 0.f ? l : ALPHA * l;
        sm[h] += __expf(l - mloc[h]);
      }
    }
  }

  for (int h = 0; h < HH; ++h) {
    red[tid] = sm[h];
    __syncthreads();
    for (int s = 128; s > 0; s >>= 1) { if (tid < s) red[tid] += red[tid + s]; __syncthreads(); }
    if (tid == 0) {
      float enm = __expf(-mloc[h]);
      float Z = (float)(NN - degsh) * enm + red[0];
      rowM[h * NN + i]    = mloc[h];
      rowENM[h * NN + i]  = enm;
      rowInvZ[h * NN + i] = 1.0f / Z;
    }
    __syncthreads();
  }
}

// ---------------- K4: fused softmax @ feats GEMM ----------------
// All 4 waves of a block share head h -> B panel featsT[h][0..63][jb:jb+32]
// is block-uniform: streamed by TDM into double-buffered LDS, overlapped
// one iteration ahead with the exp-heavy P-tile construction.
__global__ void __launch_bounds__(128) k_aggr(const float* __restrict__ A,
                                              const _Float16* __restrict__ featsT,
                                              const float* __restrict__ sSelf,
                                              const float* __restrict__ sNeigh,
                                              const float* __restrict__ rowM,
                                              const float* __restrict__ rowENM,
                                              const float* __restrict__ rowInvZ,
                                              const float* __restrict__ bias,
                                              float* __restrict__ out) {
  __shared__ __align__(16) unsigned char sB[2][BPANEL_BYTES];

  int wv   = threadIdx.x >> 5;
  int wid  = blockIdx.x * 4 + wv;
  int h    = wid >> 8;               // block-uniform (4 | 256)
  int it   = wid & 255;
  int lane = threadIdx.x & 31;
  int half = lane >> 4;
  int lr   = lane & 15;

  int   i    = it * 16 + lr;          // A-frag row this lane owns
  float si   = sSelf[h * NN + i];
  float mi   = rowM[h * NN + i];
  float enmi = rowENM[h * NN + i];
  const float* arow = A + (size_t)i * NN;
  const float* snh  = sNeigh + h * NN;
  const _Float16* bpanel = featsT + (size_t)h * FH * NN;   // [64][4096] f16

  v8f acc[4];
#pragma unroll
  for (int t = 0; t < 4; ++t)
#pragma unroll
    for (int r = 0; r < 8; ++r) acc[t][r] = 0.0f;

  // prologue: TDM fetch of first B panel
  if (wv == 0) {
    tdm_load_tile_2d((uint32_t)(uintptr_t)&sB[0][0], bpanel,
                     NN, FH, 32, FH, NN);
  }

  int cur = 0;
  for (int jb = 0; jb < NN; jb += 32, cur ^= 1) {
    if (wv == 0) {
      if (jb + 32 < NN) {
        tdm_load_tile_2d((uint32_t)(uintptr_t)&sB[cur ^ 1][0], bpanel + jb + 32,
                         NN, FH, 32, FH, NN);
        __builtin_amdgcn_s_wait_tensorcnt(1);   // panel for jb is complete
      } else {
        __builtin_amdgcn_s_wait_tensorcnt(0);
      }
    }
    __syncthreads();                            // publish sB[cur]

    if (jb + 32 < NN) __builtin_prefetch(arow + jb + 32, 0, 1);

    // build P tile (16x32 f16) in registers per ISA A-frag layout
    FragAB p;
#pragma unroll
    for (int g = 0; g < 2; ++g) {
      int k0 = jb + g * 16 + half * 8;
      v4f a0 = *(const v4f*)(arow + k0);
      v4f a1 = *(const v4f*)(arow + k0 + 4);
      v4f s0 = *(const v4f*)(snh + k0);
      v4f s1 = *(const v4f*)(snh + k0 + 4);
#pragma unroll
      for (int c = 0; c < 4; ++c) {
        float l0 = si + s0[c]; l0 = l0 > 0.f ? l0 : ALPHA * l0;
        float l1 = si + s1[c]; l1 = l1 > 0.f ? l1 : ALPHA * l1;
        float p0 = (a0[c] > 0.5f) ? __expf(l0 - mi) : enmi;   // p <= 1, f16-safe
        float p1 = (a1[c] > 0.5f) ? __expf(l1 - mi) : enmi;
        p.h[g * 8 + c]     = (_Float16)p0;
        p.h[g * 8 + 4 + c] = (_Float16)p1;
      }
    }

    // B fragments from padded LDS panel (80B row stride: bank-conflict-free)
#pragma unroll
    for (int t = 0; t < 4; ++t) {
      const unsigned char* bp = &sB[cur][(t * 16 + lr) * BPANEL_STRIDE + half * 16];
      FragAB b;
      b.q[0] = *(const v4u*)(bp);
      b.q[1] = *(const v4u*)(bp + 32);
      acc[t] = __builtin_amdgcn_wmma_f32_16x16x32_f16(
          false, p.v, false, b.v, (short)0, acc[t], false, false);
    }
    __syncthreads();                            // all readers done with sB[cur]
  }

#pragma unroll
  for (int t = 0; t < 4; ++t) {
    int n = t * 16 + lr;
    float bv = bias[h * FH + n];
#pragma unroll
    for (int r = 0; r < 8; ++r) {
      int M = it * 16 + half * 8 + r;
      float v = acc[t][r] * rowInvZ[h * NN + M] + bv;
      v = v > 0.f ? v : 0.f;
      out[(size_t)M * (HH * FH) + h * FH + n] = v;
    }
  }
}

// ---------------- launcher ----------------
extern "C" void kernel_launch(void* const* d_in, const int* in_sizes, int n_in,
                              void* d_out, int out_size, void* d_ws, size_t ws_size,
                              hipStream_t stream) {
  (void)in_sizes; (void)n_in; (void)out_size; (void)ws_size;
  const float* X       = (const float*)d_in[0];
  const float* A       = (const float*)d_in[1];
  const float* W       = (const float*)d_in[2];
  const float* a_self  = (const float*)d_in[3];
  const float* a_neigh = (const float*)d_in[4];
  const float* b       = (const float*)d_in[5];
  float* out = (float*)d_out;

  char* ws = (char*)d_ws;
  size_t off = 0;
  _Float16* Xh     = (_Float16*)(ws + off); off += (size_t)NN * FIN * 2;       // 4 MB
  _Float16* Wt     = (_Float16*)(ws + off); off += (size_t)HH * FH * FIN * 2;  // 0.5 MB
  float*    featsF = (float*)   (ws + off); off += (size_t)HH * NN * FH * 4;   // 8 MB
  _Float16* featsT = (_Float16*)(ws + off); off += (size_t)HH * FH * NN * 2;   // 4 MB
  float*    sSelf  = (float*)   (ws + off); off += (size_t)HH * NN * 4;
  float*    sNeigh = (float*)   (ws + off); off += (size_t)HH * NN * 4;
  float*    rowM   = (float*)   (ws + off); off += (size_t)HH * NN * 4;
  float*    rowIZ  = (float*)   (ws + off); off += (size_t)HH * NN * 4;
  float*    rowENM = (float*)   (ws + off); off += (size_t)HH * NN * 4;

  k_cvt_x<<<(NN * FIN) / 256, 256, 0, stream>>>(X, Xh, NN * FIN);
  k_cvt_w<<<(HH * FIN * FH) / 256, 256, 0, stream>>>(W, Wt);
  k_proj<<<(HH * (NN / 16)) / 4, 128, 0, stream>>>(Xh, Wt, featsF, featsT);
  k_scores<<<(HH * NN) / 256, 256, 0, stream>>>(featsF, a_self, a_neigh, sSelf, sNeigh);
  k_stats<<<NN, 256, 0, stream>>>(A, sSelf, sNeigh, rowM, rowIZ, rowENM);
  k_aggr<<<(HH * (NN / 16)) / 4, 128, 0, stream>>>(A, featsT, sSelf, sNeigh,
                                                   rowM, rowENM, rowIZ, b, out);
}
